// LiDARNeRF_10874857193731
// MI455X (gfx1250) — compile-verified, matrix operand order checked by reading, not run on hardware
//
#include <hip/hip_runtime.h>
#include <math.h>

typedef __attribute__((ext_vector_type(16))) _Float16 v16h;
typedef __attribute__((ext_vector_type(8)))  _Float16 v8h;
typedef __attribute__((ext_vector_type(8)))  float    v8f;

#define P_RAYS 2048
#define S_SAMP 64
#define K_NBR  8

// padded/transposed f16 weight sizes (elements)
#define N_WT0 (128*96)     // W0^T : [128][96]  (88 in-feats padded to 96)
#define N_WT1 (128*128)
#define N_WT2 (128*128)
#define N_WA0 (64*128)     // Wa0^T: [64][128]
#define N_WR0 (64*128)     // Wr0^T: [64][128]
#define N_WALL (N_WT0+N_WT1+N_WT2+N_WA0+N_WR0)   // 61440 f16

// d_ws layout
#define WS_NSAMP (P_RAYS*S_SAMP)                 // 131072
#define WS_W_OFF_BYTES 2097152                   // after alpha[131072]f32 + rgb[131072*3]f32

// dynamic LDS layout (f16 elems): weights | act[128][128] | fs[16][128] | ah[16][64] | rh[16][64]
#define SMEM_F16   (N_WALL + 128*128 + 16*128 + 16*64 + 16*64)   // 81920
#define SMEM_BYTES (SMEM_F16*2 + 128*4 + 16*4)                   // 164416

static __device__ __forceinline__ float lrelu_f(float x) { return x >= 0.f ? x : 0.1f * x; }
static __device__ __forceinline__ float sigm_f(float x)  { return 1.f / (1.f + expf(-x)); }

// Load a 16x32 f16 fragment (A-layout per CDNA5 ISA 7.12.2) from a row-major
// [rows][ld] f16 matrix in LDS. Works for both A (activations) and B
// (weights stored transposed [N][K], so "row" = output neuron).
static __device__ __forceinline__ v16h load_frag(const _Float16* base, int rowBase, int ld, int k0) {
  const int lane = threadIdx.x & 31;
  const _Float16* p = base + (rowBase + (lane & 15)) * ld + k0 + ((lane >> 4) << 3);
  v8h lo = *(const v8h*)(p);        // K = kb .. kb+7
  v8h hi = *(const v8h*)(p + 16);   // K = kb+16 .. kb+23
  v16h r;
#pragma unroll
  for (int i = 0; i < 8; ++i) { r[i] = lo[i]; r[i + 8] = hi[i]; }
  return r;
}

// ---------------- weight conversion: fp32 [K][N] -> f16 [N][Kpad] ----------------
__global__ __launch_bounds__(256) void convert_weights(
    const float* __restrict__ W0, const float* __restrict__ W1, const float* __restrict__ W2,
    const float* __restrict__ Wa0, const float* __restrict__ Wr0, _Float16* __restrict__ out) {
  int i = blockIdx.x * 256 + threadIdx.x;
  if (i >= N_WALL) return;
  float v;
  if (i < N_WT0) {
    int n = i / 96, k = i % 96;
    v = (k < 88) ? W0[k * 128 + n] : 0.f;
  } else if (i < N_WT0 + N_WT1) {
    int t = i - N_WT0; int n = t / 128, k = t % 128;
    v = W1[k * 128 + n];
  } else if (i < N_WT0 + N_WT1 + N_WT2) {
    int t = i - N_WT0 - N_WT1; int n = t / 128, k = t % 128;
    v = W2[k * 128 + n];
  } else if (i < N_WT0 + N_WT1 + N_WT2 + N_WA0) {
    int t = i - N_WT0 - N_WT1 - N_WT2; int n = t / 128, k = t % 128;
    v = Wa0[k * 64 + n];
  } else {
    int t = i - N_WT0 - N_WT1 - N_WT2 - N_WA0; int n = t / 128, k = t % 128;
    v = Wr0[k * 64 + n];
  }
  out[i] = (_Float16)v;
}

// ---------------- main WMMA kernel ----------------
__global__ __launch_bounds__(256) void nerf_main(
    const float* __restrict__ map_xyz, const float* __restrict__ map_feat,
    const int* __restrict__ ind_voxel, const float* __restrict__ sample_xyz,
    const float* __restrict__ b0g, const float* __restrict__ b1g, const float* __restrict__ b2g,
    const float* __restrict__ ba0g, const float* __restrict__ ba1g,
    const float* __restrict__ br0g, const float* __restrict__ br1g,
    const float* __restrict__ Wa1g, const float* __restrict__ Wr1g,
    const _Float16* __restrict__ wWeights,
    float* __restrict__ alphaWS, float* __restrict__ rgbWS) {
  extern __shared__ char smem_raw[];
  _Float16* sW   = (_Float16*)smem_raw;
  _Float16* sWT0 = sW;
  _Float16* sWT1 = sWT0 + N_WT0;
  _Float16* sWT2 = sWT1 + N_WT1;
  _Float16* sWa0 = sWT2 + N_WT2;
  _Float16* sWr0 = sWa0 + N_WA0;
  _Float16* sAct = sWr0 + N_WR0;        // [128][96] input, then [128][128]
  _Float16* sFS  = sAct + 128 * 128;    // [16][128] feat_s (f16)
  _Float16* sAH  = sFS + 16 * 128;      // [16][64]
  _Float16* sRH  = sAH + 16 * 64;       // [16][64]
  float* sWraw = (float*)(sRH + 16 * 64); // [128]
  float* sWsum = sWraw + 128;             // [16]

  const int tid  = threadIdx.x;
  const int wave = tid >> 5;
  const int lane = tid & 31;
  const int blockRow0 = blockIdx.x * 128;  // first flat (p,s,k) row
  const int gs0 = blockIdx.x * 16;         // first flat (p,s) sample

  // --- stage all f16 weights into LDS (120 KB) ---
  {
    const uint4* src = (const uint4*)wWeights;
    uint4* dst = (uint4*)sW;
    for (int i = tid; i < (N_WALL * 2) / 16; i += 256) dst[i] = src[i];
  }

  // --- build input activations [128][96]: gathered feats + posenc + pad ---
  {
    int r = tid >> 1, half = tid & 1;
    int gr = blockRow0 + r;
    int vid = ind_voxel[gr];
    const float4* f4 = (const float4*)(map_feat + (size_t)vid * 64 + half * 32);
    _Float16* d = sAct + r * 96 + half * 32;
#pragma unroll
    for (int i = 0; i < 8; ++i) {
      float4 v = f4[i];
      d[i * 4 + 0] = (_Float16)v.x; d[i * 4 + 1] = (_Float16)v.y;
      d[i * 4 + 2] = (_Float16)v.z; d[i * 4 + 3] = (_Float16)v.w;
    }
  }
  if (tid < 128) {
    int r = tid;
    int gr = blockRow0 + r;
    int gs = gr >> 3;
    int vid = ind_voxel[gr];
    float o[3];
#pragma unroll
    for (int dd = 0; dd < 3; ++dd)
      o[dd] = sample_xyz[gs * 3 + dd] - map_xyz[(size_t)vid * 3 + dd];
    _Float16* d = sAct + r * 96 + 64;
#pragma unroll
    for (int dd = 0; dd < 3; ++dd) {
      float f = 1.f;
#pragma unroll
      for (int l = 0; l < 4; ++l) {
        float ang = o[dd] * f;
        d[dd * 8 + l]     = (_Float16)sinf(ang);
        d[dd * 8 + 4 + l] = (_Float16)cosf(ang);
        f *= 2.f;
      }
    }
#pragma unroll
    for (int i = 0; i < 8; ++i) d[24 + i] = (_Float16)0.f;
    sWraw[r] = expf(-10.f * sqrtf(o[0] * o[0] + o[1] * o[1] + o[2] * o[2]));
  }
  __syncthreads();

  // --- one MLP layer: act[128][kdim] @ Wt^T -> lrelu -> act[128][128] ---
  // Wave tiling: 4x2 grid. Each wave owns 2 M-subtiles x 4 N-tiles, so per
  // layer it issues 8 A-frag + 16 B-frag LDS loads for 32 WMMAs (vs 36 for a
  // 1x8 arrangement) -- minimizes LDS feeder bandwidth per matrix op.
  auto layer = [&](const _Float16* wt, const float* bias, int kdim) {
    const int nk = kdim >> 5;
    const int m0 = (wave & 3) * 32;      // two 16-row subtiles: m0, m0+16
    const int n0 = (wave >> 2) * 64;     // four 16-col tiles:   n0 + j*16
    v16h afr[2][4];
#pragma unroll
    for (int mm = 0; mm < 2; ++mm)
      for (int kc = 0; kc < nk; ++kc)
        afr[mm][kc] = load_frag(sAct, m0 + mm * 16, kdim, kc * 32);
    v8f acc[2][4];
#pragma unroll
    for (int mm = 0; mm < 2; ++mm)
#pragma unroll
      for (int j = 0; j < 4; ++j)
#pragma unroll
        for (int e = 0; e < 8; ++e) acc[mm][j][e] = 0.f;
#pragma unroll
    for (int j = 0; j < 4; ++j)
      for (int kc = 0; kc < nk; ++kc) {
        v16h b = load_frag(wt, n0 + j * 16, kdim, kc * 32);
        acc[0][j] = __builtin_amdgcn_wmma_f32_16x16x32_f16(
            false, afr[0][kc], false, b, (short)0, acc[0][j], false, false);
        acc[1][j] = __builtin_amdgcn_wmma_f32_16x16x32_f16(
            false, afr[1][kc], false, b, (short)0, acc[1][j], false, false);
      }
    __syncthreads();
    const int col = lane & 15, rOff = (lane >> 4) * 8;
#pragma unroll
    for (int j = 0; j < 4; ++j) {
      float bj = bias[n0 + j * 16 + col];
#pragma unroll
      for (int mm = 0; mm < 2; ++mm)
#pragma unroll
        for (int i = 0; i < 8; ++i)
          sAct[(m0 + mm * 16 + rOff + i) * 128 + n0 + j * 16 + col] =
              (_Float16)lrelu_f(acc[mm][j][i] + bj);
    }
    __syncthreads();
  };

  layer(sWT0, b0g, 96);
  layer(sWT1, b1g, 128);
  layer(sWT2, b2g, 128);

  // --- normalized distance weights, reduce K=8 neighbors -> feat_s[16][128] ---
  if (tid < 16) {
    float s = 0.f;
#pragma unroll
    for (int k = 0; k < 8; ++k) s += sWraw[tid * 8 + k];
    sWsum[tid] = s;
  }
  __syncthreads();
  if (tid < 128) sWraw[tid] = sWraw[tid] / sWsum[tid >> 3];
  __syncthreads();
  {
    int c = tid & 127;
    int ls0 = (tid >> 7) * 8;
    for (int u = 0; u < 8; ++u) {
      int ls = ls0 + u;
      float s = 0.f;
#pragma unroll
      for (int k = 0; k < 8; ++k)
        s += (float)sAct[(ls * 8 + k) * 128 + c] * sWraw[ls * 8 + k];
      sFS[ls * 128 + c] = (_Float16)s;
    }
  }
  __syncthreads();

  // --- heads via WMMA: waves 0-3 alpha-head tiles, waves 4-7 rgb-head tiles ---
  {
    const _Float16* wt  = (wave < 4) ? sWa0 : sWr0;
    const float* bias   = (wave < 4) ? ba0g : br0g;
    _Float16* dsth      = (wave < 4) ? sAH : sRH;
    const int j = wave & 3;
    v8f acc;
#pragma unroll
    for (int e = 0; e < 8; ++e) acc[e] = 0.f;
#pragma unroll
    for (int kc = 0; kc < 4; ++kc) {
      v16h a = load_frag(sFS, 0, 128, kc * 32);
      v16h b = load_frag(wt, j * 16, 128, kc * 32);
      acc = __builtin_amdgcn_wmma_f32_16x16x32_f16(
          false, a, false, b, (short)0, acc, false, false);
    }
    const int col = lane & 15, rOff = (lane >> 4) * 8;
    float bj = bias[j * 16 + col];
#pragma unroll
    for (int i = 0; i < 8; ++i)
      dsth[(rOff + i) * 64 + j * 16 + col] = (_Float16)lrelu_f(acc[i] + bj);
  }
  __syncthreads();

  // --- tiny final layers + sigmoid; write per-sample alpha/rgb to workspace ---
  if (tid < 16) {
    int ls = tid;
    float s = ba1g[0];
    for (int i = 0; i < 64; ++i) s += (float)sAH[ls * 64 + i] * Wa1g[i];
    alphaWS[gs0 + ls] = sigm_f(s);
  }
  if (tid >= 64 && tid < 112) {
    int u = tid - 64;
    int ls = u / 3, ch = u % 3;
    float s = br1g[ch];
    for (int i = 0; i < 64; ++i) s += (float)sRH[ls * 64 + i] * Wr1g[i * 3 + ch];
    rgbWS[(gs0 + ls) * 3 + ch] = sigm_f(s);
  }
}

// ---------------- volume rendering: one thread per ray ----------------
__global__ __launch_bounds__(256) void render_rays(
    const float* __restrict__ alphaWS, const float* __restrict__ rgbWS,
    const float* __restrict__ dist_to_cam, float* __restrict__ out) {
  int p = blockIdx.x * 256 + threadIdx.x;
  if (p >= P_RAYS) return;
  float T = 1.f, r = 0.f, g = 0.f, b = 0.f, d = 0.f, acc = 0.f;
  for (int s = 0; s < S_SAMP; ++s) {
    int idx = p * S_SAMP + s;
    float a = alphaWS[idx];
    float dist = 1.0f + 80.0f * dist_to_cam[idx] * (1.0f / 64.0f);
    float blend = a * T;
    r += rgbWS[idx * 3 + 0] * blend;
    g += rgbWS[idx * 3 + 1] * blend;
    b += rgbWS[idx * 3 + 2] * blend;
    d += dist * blend;
    acc += blend;
    T *= (1.0f - a + 1e-10f);
  }
  out[p * 5 + 0] = r; out[p * 5 + 1] = g; out[p * 5 + 2] = b;
  out[p * 5 + 3] = d; out[p * 5 + 4] = acc;
}

extern "C" void kernel_launch(void* const* d_in, const int* in_sizes, int n_in,
                              void* d_out, int out_size, void* d_ws, size_t ws_size,
                              hipStream_t stream) {
  (void)in_sizes; (void)n_in; (void)out_size; (void)ws_size;
  const float* map_xyz     = (const float*)d_in[0];
  const float* map_feat    = (const float*)d_in[1];
  const int*   ind_voxel   = (const int*)d_in[2];
  const float* sample_xyz  = (const float*)d_in[3];
  const float* dist_to_cam = (const float*)d_in[4];
  const float* W0 = (const float*)d_in[5];  const float* b0 = (const float*)d_in[6];
  const float* W1 = (const float*)d_in[7];  const float* b1 = (const float*)d_in[8];
  const float* W2 = (const float*)d_in[9];  const float* b2 = (const float*)d_in[10];
  const float* Wa0 = (const float*)d_in[11]; const float* ba0 = (const float*)d_in[12];
  const float* Wa1 = (const float*)d_in[13]; const float* ba1 = (const float*)d_in[14];
  const float* Wr0 = (const float*)d_in[15]; const float* br0 = (const float*)d_in[16];
  const float* Wr1 = (const float*)d_in[17]; const float* br1 = (const float*)d_in[18];

  float* alphaWS = (float*)d_ws;
  float* rgbWS   = alphaWS + WS_NSAMP;
  _Float16* wW   = (_Float16*)((char*)d_ws + WS_W_OFF_BYTES);

  convert_weights<<<(N_WALL + 255) / 256, 256, 0, stream>>>(W0, W1, W2, Wa0, Wr0, wW);

  hipFuncSetAttribute((const void*)nerf_main,
                      hipFuncAttributeMaxDynamicSharedMemorySize, SMEM_BYTES);
  nerf_main<<<(P_RAYS * S_SAMP * K_NBR) / 128, 256, SMEM_BYTES, stream>>>(
      map_xyz, map_feat, ind_voxel, sample_xyz,
      b0, b1, b2, ba0, ba1, br0, br1, Wa1, Wr1,
      wW, alphaWS, rgbWS);

  render_rays<<<(P_RAYS + 255) / 256, 256, 0, stream>>>(alphaWS, rgbWS, dist_to_cam, (float*)d_out);
}
